// AutoCorrelation_91044716740872
// MI455X (gfx1250) — compile-verified, matrix operand order checked by reading, not run on hardware
//
#include <hip/hip_runtime.h>
#include <hip/hip_bf16.h>
#include <math.h>

// ---------------------------------------------------------------------------
// AutoCorrelation (Autoformer) for MI455X / gfx1250.
// DFT-as-GEMM formulation: three f16-in/f32-acc WMMA GEMMs against shared
// (pre-transposed) DFT matrices, wave32 top-k+softmax, LDS-staged circular
// aggregation. GEMM uses GLOBAL_LOAD_ASYNC_TO_LDS_B128 double-buffered
// staging (ASYNCcnt) so HBM/L2 latency hides behind v_wmma issue.
//
//   B=16, L=2048, H=16, E=64  ->  C = B*H*E = 16384 channels of length 2048.
//
// Workspace (~288 MB):
//   Aq  @0      16384x2048 f16 | Ak  @64M  16384x2048 f16
//   BfT @128M    2048x2048 f16 | BiT @136M  2048x2048 f16   (row n/tau, col t/r)
//   QF  @144M  16384x2048 f16 | KF  @208M  16384x2048 f16
//   W   @272M  16384x16 f32   | Dn  @273M  16384x16 i32
//   P reuses Aq, corr reuses Ak.
// ---------------------------------------------------------------------------

typedef __attribute__((ext_vector_type(16))) _Float16 v16h;
typedef __attribute__((ext_vector_type(8)))  float    v8f;
typedef __attribute__((ext_vector_type(4)))  int      v4i_;

#define BB 16
#define LL 2048
#define HH 16
#define EE 64
#define CH (BB * HH * EE)   // 16384
#define TOPK 15
#define EG 8
#define KSTEP 64

// ---- gfx1250 async global->LDS helpers ------------------------------------
#if defined(__has_builtin)
#if __has_builtin(__builtin_amdgcn_global_load_async_to_lds_b128)
#define HAS_ASYNC_BUILTIN 1
#endif
#if __has_builtin(__builtin_amdgcn_s_wait_asynccnt)
#define HAS_WAIT_BUILTIN 1
#endif
#endif

__device__ __forceinline__ void async_b128(const _Float16* g, _Float16* l) {
#ifdef HAS_ASYNC_BUILTIN
  // prototype (from hipcc diagnostic): void(v4i AS1*, v4i AS3*, Imm, Imm)
  __builtin_amdgcn_global_load_async_to_lds_b128(
      (__attribute__((address_space(1))) v4i_*)(_Float16*)g,
      (__attribute__((address_space(3))) v4i_*)l, 0, 0);
#else
  unsigned loff = (unsigned)(unsigned long long)
      (__attribute__((address_space(3))) void*)l;
  unsigned long long ga = (unsigned long long)g;
  asm volatile("global_load_async_to_lds_b128 %0, %1, off"
               :: "v"(loff), "v"(ga) : "memory");
#endif
}
__device__ __forceinline__ void async_wait8() {
#ifdef HAS_WAIT_BUILTIN
  __builtin_amdgcn_s_wait_asynccnt(8);
#else
  asm volatile("s_wait_asynccnt 0x8" ::: "memory");
#endif
}
__device__ __forceinline__ void async_wait0() {
#ifdef HAS_WAIT_BUILTIN
  __builtin_amdgcn_s_wait_asynccnt(0);
#else
  asm volatile("s_wait_asynccnt 0x0" ::: "memory");
#endif
}

// ---------------------------------------------------------------------------
// Pre-TRANSPOSED DFT matrices in f16 (row = output index, col = reduction).
//  BfT[n][t]  : n<1024 -> cos(2*pi*n*t/L) (n==0 zeroed = band kills DC)
//               n>=1024 -> sin(2*pi*(n-1024)*t/L)
//  BiT[tau][r]: r<1024 -> cos(2*pi*r*tau/L) ; r>=1024 -> -sin(2*pi*(r-1024)*tau/L)
// ---------------------------------------------------------------------------
__global__ __launch_bounds__(256) void ac_gen_dft(_Float16* __restrict__ BfT,
                                                  _Float16* __restrict__ BiT) {
  size_t gid = (size_t)blockIdx.x * 256 + threadIdx.x;   // 0 .. 2*2^22-1
  int which = (int)(gid >> 22);
  int idx   = (int)(gid & ((1u << 22) - 1));
  int row = idx >> 11;
  int col = idx & (LL - 1);
  const float PI_OVER_1024 = 3.0679615757712823e-3f;
  if (which == 0) {
    int f = row & 1023;
    bool is_sin = row >= 1024;
    int a = (f * col) & (LL - 1);
    float th = (float)a * PI_OVER_1024;
    float val = is_sin ? __sinf(th) : __cosf(th);
    if (f == 0) val = 0.0f;
    BfT[idx] = (_Float16)val;
  } else {
    int f = col & 1023;
    bool is_sin = col >= 1024;
    int a = (f * row) & (LL - 1);
    float th = (float)a * PI_OVER_1024;
    float val = is_sin ? -__sinf(th) : __cosf(th);
    BiT[idx] = (_Float16)val;
  }
}

// ---------------------------------------------------------------------------
// Pack [B,L,H*E] f32 -> [B*H*E, L] f16 (tiled 32x32 transpose through LDS).
// ---------------------------------------------------------------------------
__global__ __launch_bounds__(256) void ac_pack(const float* __restrict__ src,
                                               _Float16* __restrict__ dst) {
  __shared__ float tile[32][33];
  int b  = blockIdx.z;
  int t0 = blockIdx.x * 32;
  int c0 = blockIdx.y * 32;
  int tx = threadIdx.x;
  int ty = threadIdx.y;
  #pragma unroll
  for (int i = 0; i < 32; i += 8) {
    int t = t0 + ty + i;
    tile[ty + i][tx] = src[((size_t)b * LL + t) * (HH * EE) + c0 + tx];
  }
  __syncthreads();
  #pragma unroll
  for (int i = 0; i < 32; i += 8) {
    int c = c0 + ty + i;
    dst[((size_t)b * (HH * EE) + c) * LL + t0 + tx] = (_Float16)tile[tx][ty + i];
  }
}

// ---------------------------------------------------------------------------
// D[m][n] = sum_k A[m][k] * Bt[n][k]   (both row-major, f16, f32 accumulate)
// Workgroup tile 128x128, 8 waves (2M x 4N), each wave 4x2 WMMA tiles.
// K staged 64 at a time, double-buffered, via async global->LDS b128:
// per wave per stage: 8 async instructions (4 A + 4 Bt), each 32 lanes x 16B.
// ---------------------------------------------------------------------------
__global__ __launch_bounds__(256) void ac_gemm(const _Float16* __restrict__ A,
                                               const _Float16* __restrict__ Bt,
                                               _Float16* __restrict__ D,
                                               int M, int N, int K) {
  __shared__ _Float16 sA[2][128 * KSTEP];   // 2 x 16 KB
  __shared__ _Float16 sB[2][128 * KSTEP];   // 2 x 16 KB
  int tid  = threadIdx.x;
  int wave = tid >> 5;
  int lane = tid & 31;
  int wm = wave >> 2;                  // 0..1
  int wn = wave & 3;                   // 0..3
  int m0 = blockIdx.y * 128;
  int n0 = blockIdx.x * 128;
  int lg = lane >> 4;
  int lm = lane & 15;

  // WMMA 16-bit A-operand K mapping (VGPR r holds half-pair at K0)
  int kmap[8];
  #pragma unroll
  for (int r = 0; r < 8; ++r)
    kmap[r] = (r < 4 ? 2 * r : 8 + 2 * r) + 8 * lg;

  // staging geometry: inst g (0..31 per matrix) covers 512B = 4 rows x 128B;
  // lane covers row g*4+(lane>>3), half-col (lane&7)*8; LDS off = g*512+lane*16
  int srow = lane >> 3;
  int scol = (lane & 7) * 8;
  int soff = lane * 16;

  auto stage = [&](int buf, int k0) {
    #pragma unroll
    for (int i = 0; i < 4; ++i) {
      int g   = wave * 4 + i;
      int row = g * 4 + srow;
      const _Float16* gA = &A [(size_t)(m0 + row) * K + k0 + scol];
      const _Float16* gB = &Bt[(size_t)(n0 + row) * K + k0 + scol];
      _Float16* lA = (_Float16*)((char*)&sA[buf][0] + g * 512 + soff);
      _Float16* lB = (_Float16*)((char*)&sB[buf][0] + g * 512 + soff);
      async_b128(gA, lA);
      async_b128(gB, lB);
    }
  };

  v8f acc[4][2] = {};

  int nstep = K / KSTEP;
  stage(0, 0);
  for (int s = 0; s < nstep; ++s) {
    int cur = s & 1;
    if (s + 1 < nstep) {
      stage(cur ^ 1, (s + 1) * KSTEP);
      async_wait8();            // tile `cur` (8 older per-wave asyncs) landed
    } else {
      async_wait0();
    }
    __syncthreads();            // all waves' tile-cur LDS writes visible

    #pragma unroll
    for (int ksub = 0; ksub < 2; ++ksub) {
      v16h af[4], bf[2];
      #pragma unroll
      for (int i = 0; i < 4; ++i) {
        int ra = (wm * 64 + i * 16 + lm) * KSTEP + ksub * 32;
        #pragma unroll
        for (int r = 0; r < 8; ++r) {
          af[i][2 * r]     = sA[cur][ra + kmap[r]];
          af[i][2 * r + 1] = sA[cur][ra + kmap[r] + 1];
        }
      }
      #pragma unroll
      for (int j = 0; j < 2; ++j) {
        int rb = (wn * 32 + j * 16 + lm) * KSTEP + ksub * 32;
        #pragma unroll
        for (int r = 0; r < 8; ++r) {
          bf[j][2 * r]     = sB[cur][rb + kmap[r]];
          bf[j][2 * r + 1] = sB[cur][rb + kmap[r] + 1];
        }
      }
      #pragma unroll
      for (int i = 0; i < 4; ++i)
        #pragma unroll
        for (int j = 0; j < 2; ++j)
          acc[i][j] = __builtin_amdgcn_wmma_f32_16x16x32_f16(
              false, af[i], false, bf[j], (short)0, acc[i][j], false, false);
    }
    __syncthreads();            // done reading cur before it is re-staged
  }

  // C/D layout: VGPR p -> M = p + 8*lg, N = lane%16
  #pragma unroll
  for (int i = 0; i < 4; ++i) {
    int mBase = m0 + wm * 64 + i * 16;
    #pragma unroll
    for (int j = 0; j < 2; ++j) {
      int nCol = n0 + wn * 32 + j * 16 + lm;
      #pragma unroll
      for (int p = 0; p < 8; ++p) {
        int row = mBase + p + 8 * lg;
        D[(size_t)row * N + nCol] = (_Float16)acc[i][j][p];
      }
    }
  }
}

// ---------------------------------------------------------------------------
// P = (2/L) * Q * conj(K); packed per channel [0..1023]=Pr, [1024..2047]=Pi.
// Qr=Qc, Qi=-Qs => Pr = QcKc + QsKs ; Pi = QcKs - QsKc.
// ---------------------------------------------------------------------------
__global__ __launch_bounds__(256) void ac_pointwise(const _Float16* __restrict__ QF,
                                                    const _Float16* __restrict__ KF,
                                                    _Float16* __restrict__ P) {
  size_t gid = (size_t)blockIdx.x * 256 + threadIdx.x;
  size_t c = gid >> 10;
  int f = (int)(gid & 1023);
  size_t base = c << 11;
  float qc = (float)QF[base + f], qs = (float)QF[base + 1024 + f];
  float kc = (float)KF[base + f], ks = (float)KF[base + 1024 + f];
  const float s = 1.0f / 1024.0f;   // 2/L
  P[base + f]        = (_Float16)((qc * kc + qs * ks) * s);
  P[base + 1024 + f] = (_Float16)((qc * ks - qs * kc) * s);
}

// ---------------------------------------------------------------------------
// Top-15 + softmax per channel; one wave32 per channel, 64 vals/lane in VGPRs,
// 15 masked argmax passes via __shfl_xor butterflies.
// ---------------------------------------------------------------------------
__global__ __launch_bounds__(256) void ac_topk(const _Float16* __restrict__ corr,
                                               float* __restrict__ W,
                                               int* __restrict__ Dn) {
  int wave = threadIdx.x >> 5, lane = threadIdx.x & 31;
  int c = blockIdx.x * 8 + wave;
  const _Float16* src = corr + (size_t)c * LL;
  float v[64];
  #pragma unroll
  for (int j = 0; j < 64; ++j) v[j] = (float)src[lane + 32 * j];

  float wsel[TOPK];
  int   dsel[TOPK];
  for (int i = 0; i < TOPK; ++i) {
    float bv = -1e30f; int bj = 0;
    #pragma unroll
    for (int j = 0; j < 64; ++j)
      if (v[j] > bv) { bv = v[j]; bj = j; }
    int bidx = lane + 32 * bj;
    #pragma unroll
    for (int off = 16; off > 0; off >>= 1) {
      float ov = __shfl_xor(bv, off);
      int   oi = __shfl_xor(bidx, off);
      if (ov > bv) { bv = ov; bidx = oi; }
    }
    wsel[i] = bv; dsel[i] = bidx;
    if (lane == (bidx & 31)) v[bidx >> 5] = -1e30f;
  }
  if (lane == 0) {
    float mx = wsel[0], sum = 0.0f, e[TOPK];
    #pragma unroll
    for (int i = 0; i < TOPK; ++i) { e[i] = __expf(wsel[i] - mx); sum += e[i]; }
    float inv = 1.0f / sum;
    #pragma unroll
    for (int i = 0; i < TOPK; ++i) {
      W[(size_t)c * 16 + i]  = e[i] * inv;
      Dn[(size_t)c * 16 + i] = dsel[i];
    }
  }
}

// ---------------------------------------------------------------------------
// out[b,t,h,e] = sum_i w_i * v[b,(t+d_i)%L,h,e] ; v staged once in 64KB LDS
// so the 15 circular gathers stay on-chip.
// ---------------------------------------------------------------------------
__global__ __launch_bounds__(256) void ac_aggregate(const float* __restrict__ v,
                                                    const float* __restrict__ W,
                                                    const int* __restrict__ Dn,
                                                    float* __restrict__ out) {
  __shared__ float vt[EG * LL];     // 64 KB
  int blk = blockIdx.x;
  int eg = blk & 7;
  int h  = (blk >> 3) & (HH - 1);
  int b  = blk >> 7;
  int e0 = eg * EG;
  int tid = threadIdx.x;

  for (int idx = tid; idx < EG * LL; idx += 256) {
    int e = idx & (EG - 1);
    int t = idx >> 3;
    vt[e * LL + t] = v[(((size_t)b * LL + t) * HH + h) * EE + e0 + e];
  }
  __syncthreads();

  for (int e = 0; e < EG; ++e) {
    int c = (b * HH + h) * EE + e0 + e;
    float w[TOPK]; int d[TOPK];
    #pragma unroll
    for (int i = 0; i < TOPK; ++i) {
      w[i] = W[(size_t)c * 16 + i];
      d[i] = Dn[(size_t)c * 16 + i];
    }
    for (int t = tid; t < LL; t += 256) {
      float acc = 0.0f;
      #pragma unroll
      for (int i = 0; i < TOPK; ++i)
        acc += w[i] * vt[e * LL + ((t + d[i]) & (LL - 1))];
      out[(((size_t)b * LL + t) * HH + h) * EE + e0 + e] = acc;
    }
  }
}

// ---------------------------------------------------------------------------
extern "C" void kernel_launch(void* const* d_in, const int* in_sizes, int n_in,
                              void* d_out, int out_size, void* d_ws, size_t ws_size,
                              hipStream_t stream) {
  const float* q = (const float*)d_in[0];
  const float* k = (const float*)d_in[1];
  const float* v = (const float*)d_in[2];
  float* out = (float*)d_out;

  char* ws = (char*)d_ws;   // requires ~288 MB of workspace
  _Float16* Aq  = (_Float16*)(ws + 0ull);
  _Float16* Ak  = (_Float16*)(ws + 67108864ull);
  _Float16* BfT = (_Float16*)(ws + 134217728ull);
  _Float16* BiT = (_Float16*)(ws + 142606336ull);
  _Float16* QF  = (_Float16*)(ws + 150994944ull);
  _Float16* KF  = (_Float16*)(ws + 218103808ull);
  float*    W   = (float*)   (ws + 285212672ull);
  int*      Dn  = (int*)     (ws + 286261248ull);
  _Float16* P    = Aq;   // Aq dead after GEMM1(q)
  _Float16* corr = Ak;   // Ak dead after GEMM1(k)

  // 1) pre-transposed DFT matrices (f16, L2-resident, reused by all channels)
  ac_gen_dft<<<32768, 256, 0, stream>>>(BfT, BiT);

  // 2) pack q,k: [B,L,H,E] f32 -> [C,L] f16
  dim3 pg(LL / 32, (HH * EE) / 32, BB), pb(32, 8);
  ac_pack<<<pg, pb, 0, stream>>>(q, Aq);
  ac_pack<<<pg, pb, 0, stream>>>(k, Ak);

  // 3) forward DFTs as GEMMs: [16384x2048] x [2048x2048]^T-layout
  dim3 gg(2048 / 128, CH / 128);
  ac_gemm<<<gg, 256, 0, stream>>>(Aq, BfT, QF, CH, LL, LL);
  ac_gemm<<<gg, 256, 0, stream>>>(Ak, BfT, KF, CH, LL, LL);

  // 4) spectrum product with 2/L scale (band baked into DFT rows)
  ac_pointwise<<<(CH * 1024) / 256, 256, 0, stream>>>(QF, KF, P);

  // 5) inverse DFT as GEMM -> corr
  ac_gemm<<<gg, 256, 0, stream>>>(P, BiT, corr, CH, LL, LL);

  // 6) top-15 + softmax per channel
  ac_topk<<<CH / 8, 256, 0, stream>>>(corr, W, Dn);

  // 7) LDS-staged circular-shift aggregation + output transpose
  ac_aggregate<<<BB * HH * (EE / EG), 256, 0, stream>>>(v, W, Dn, out);
}
